// TestBNNmodel_68229850464342
// MI455X (gfx1250) — matrix operand order repaired and checked by reading, not compiled, not run on hardware
//
#include <hip/hip_runtime.h>
#include <hip/hip_bf16.h>

typedef __attribute__((ext_vector_type(16))) _Float16 v16h;
typedef __attribute__((ext_vector_type(8)))  float    v8f;
typedef _Float16 half_t;

#define N_SAMP   96
#define S_PTS    4096
#define P_DIM    4096
#define PH_DIM   1024
#define PHH_DIM  512
#define TP       3267          // total hypernet-generated params per sample
#define NT_B     205           // ceil(3267/16)
#define W0_F     30.0f

__device__ __forceinline__ float st_f(float x) {
    // softplus(x)/6, numerically stable
    float sp = fmaxf(x, 0.0f) + log1pf(__expf(-fabsf(x)));
    return sp * (1.0f / 6.0f);
}

// ---------------------------------------------------------------------------
// Kernel 1: z[n,p] = loc + eps*st(log_scale) + hs gather + hhs gather  -> f16
// ---------------------------------------------------------------------------
__global__ __launch_bounds__(256)
void z_kernel(const float* __restrict__ loc, const float* __restrict__ log_scale,
              const float* __restrict__ h_loc, const float* __restrict__ h_log_scale,
              const float* __restrict__ hh_loc, const float* __restrict__ hh_log_scale,
              const float* __restrict__ eps, const float* __restrict__ h_eps,
              const float* __restrict__ hh_eps,
              const int* __restrict__ h_gidx, const int* __restrict__ hh_gidx,
              half_t* __restrict__ z_h)
{
    int i = blockIdx.x * blockDim.x + threadIdx.x;
    if (i >= N_SAMP * P_DIM) return;
    int n = i >> 12;            // / 4096
    int p = i & (P_DIM - 1);

    float s = loc[i] + eps[i] * st_f(log_scale[i]);

    int g   = n >> 4;                       // n/16
    int hio = g * PH_DIM + h_gidx[p];
    float hs = h_loc[hio] + h_eps[hio] * st_f(h_log_scale[hio]);

    int hhi = hh_gidx[p];                   // n/96 == 0 always (n < 96)
    float hhs = hh_loc[hhi] + hh_eps[hhi] * st_f(hh_log_scale[hhi]);

    z_h[i] = (half_t)(s + hs + hhs);
}

// ---------------------------------------------------------------------------
// Kernel 2: params(96x3267) = z(96x4096) @ W_map(4096x3267) + b_map
// One wave per 16x16 output tile, K-loop step 32, v_wmma_f32_16x16x32_f16.
// All loads unconditional (clamped column) -> straight-line K loop.
// ---------------------------------------------------------------------------
__global__ __launch_bounds__(128)
void params_gemm_kernel(const half_t* __restrict__ z_h,
                        const float* __restrict__ W_map,
                        const float* __restrict__ b_map,
                        float* __restrict__ params)
{
    int lane = threadIdx.x & 31;
    int wid  = blockIdx.x * 4 + (threadIdx.x >> 5);
    int mt   = wid % 6;
    int nt   = wid / 6;
    if (nt >= NT_B) return;                 // whole-wave exit; EXEC stays full

    int half_hi = lane >> 4;                // 0: lanes 0-15, 1: lanes 16-31
    int l16     = lane & 15;

    int m = mt * 16 + l16;                  // A row this lane supplies (m < 96)
    const half_t* zrow = z_h + (size_t)m * P_DIM;

    int  ncol   = nt * 16 + l16;            // B/D column this lane supplies
    bool nvalid = ncol < TP;
    int  ncl    = nvalid ? ncol : (TP - 1); // clamp: loads always in-bounds;
                                            // invalid columns are never stored

    float bias = b_map[ncl];
    v8f c;
#pragma unroll
    for (int r = 0; r < 8; ++r) c[r] = bias;   // fold bias into accumulator

    for (int k0 = 0; k0 < P_DIM; k0 += 32) {
        // A fragment (16-bit 16x32 layout): two contiguous 16B half-groups
        v16h a;
        int ka0 = k0 + half_hi * 16;
        int ka1 = k0 + half_hi * 16 + 8;
#pragma unroll
        for (int j = 0; j < 8; ++j) {
            a[j]     = zrow[ka0 + j];
            a[8 + j] = zrow[ka1 + j];
        }
        // B fragment (32x16): lanes<16 K=k0..k0+15, lanes>=16 K=k0+16..k0+31
        // unconditional loads; coalesced across the 16 lanes of each half
        v16h b;
        int kb = k0 + half_hi * 16;
#pragma unroll
        for (int j = 0; j < 16; ++j)
            b[j] = (half_t)W_map[(size_t)(kb + j) * TP + ncl];

        c = __builtin_amdgcn_wmma_f32_16x16x32_f16(false, a, false, b,
                                                   (short)0, c, false, false);
    }
    if (nvalid) {
#pragma unroll
        for (int r = 0; r < 8; ++r) {
            int mm = mt * 16 + half_hi * 8 + r;   // D: VGPR r -> row half_hi*8+r
            params[(size_t)mm * TP + ncol] = c[r];
        }
    }
}

// ---------------------------------------------------------------------------
// Kernel 3: per-sample SIREN MLP 32->32->32->32->3, sin(30x) on hidden layers.
// Block = 1 sample x 128 points (8 waves x 16-point tiles).
// Layer weights stored TRANSPOSED in LDS ([dout][din]) so every B-fragment
// read is 16 contiguous, 16B-aligned halves (merges into ds_load_b128).
// D->A re-layout via per-wave LDS staging tile (pad 40 -> aligned +
// bank-conflict-free).
// ---------------------------------------------------------------------------
__global__ __launch_bounds__(256)
void mlp_kernel(const float* __restrict__ x,
                const float* __restrict__ params,
                float* __restrict__ out)
{
    __shared__ half_t WT[3][32][32];        // hidden weights, [layer][dout][din]
    __shared__ half_t WT4[3][32];           // final weights,  [dout=3][din=32]
    __shared__ half_t Bias[4][32];          // biases (layer 3 uses 3)
    __shared__ half_t stage[8][16][40];     // per-wave 16x32 staging (+pad)

    int n    = blockIdx.y;                  // sample
    int blk  = blockIdx.x;                  // point block (0..31)
    int tid  = threadIdx.x;
    int wave = tid >> 5;
    int lane = tid & 31;
    int half_hi = lane >> 4;
    int l16     = lane & 15;

    // -------- cooperative fill of LDS (transposed weight layout) --------
    const float* pn = params + (size_t)n * TP;
    for (int i = tid; i < TP; i += 256) {
        float v = pn[i];
        int layer = i / 1056;               // 1056 = 32*32+32 (layers 0..2)
        if (layer < 3) {
            int off = i - layer * 1056;
            if (off < 32) {
                Bias[layer][off] = (half_t)v;
            } else {
                int w = off - 32;           // w = k*32 + j
                int k = w >> 5, j = w & 31;
                WT[layer][j][k] = (half_t)v;
            }
        } else {
            int off = i - 3168;
            if (off < 3) {
                Bias[3][off] = (half_t)v;
            } else {
                int w = off - 3;            // w = k*3 + j
                int k = w / 3, j = w - k * 3;
                WT4[j][k] = (half_t)v;
            }
        }
    }
    __syncthreads();

    int pt_base = blk * 128 + wave * 16;    // point offset within sample

    // Load A fragment from x (row = point l16 of this tile)
    const float* xrow = x + ((size_t)n * S_PTS + pt_base + l16) * 32;
    v16h a;
    {
        int c0 = half_hi * 16;
        int c1 = half_hi * 16 + 8;
#pragma unroll
        for (int j = 0; j < 8; ++j) {
            a[j]     = (half_t)xrow[c0 + j];
            a[8 + j] = (half_t)xrow[c1 + j];
        }
    }

    // 3 hidden layers: h = sin(30*(h@W + b))
#pragma unroll
    for (int layer = 0; layer < 3; ++layer) {
        // B fragments: lane column = l16 (b0) / 16+l16 (b1);
        // K rows kb..kb+15 are CONTIGUOUS in the transposed layout.
        v16h b0, b1;
        int kb = half_hi * 16;
        const half_t* w0 = &WT[layer][l16][kb];
        const half_t* w1 = &WT[layer][16 + l16][kb];
#pragma unroll
        for (int j = 0; j < 16; ++j) {
            b0[j] = w0[j];
            b1[j] = w1[j];
        }
        float bias0 = (float)Bias[layer][l16];
        float bias1 = (float)Bias[layer][16 + l16];
        v8f c0, c1;
#pragma unroll
        for (int r = 0; r < 8; ++r) { c0[r] = bias0; c1[r] = bias1; }

        c0 = __builtin_amdgcn_wmma_f32_16x16x32_f16(false, a, false, b0,
                                                    (short)0, c0, false, false);
        c1 = __builtin_amdgcn_wmma_f32_16x16x32_f16(false, a, false, b1,
                                                    (short)0, c1, false, false);

        // activation + stage (D layout: VGPR r -> row half_hi*8+r, col l16/l16+16)
#pragma unroll
        for (int r = 0; r < 8; ++r) {
            int row = half_hi * 8 + r;
            stage[wave][row][l16]      = (half_t)__sinf(W0_F * c0[r]);
            stage[wave][row][16 + l16] = (half_t)__sinf(W0_F * c1[r]);
        }
        __syncthreads();
        // reload into A layout (two contiguous 8-half groups, 16B aligned)
        {
            int cc0 = half_hi * 16;
            int cc1 = half_hi * 16 + 8;
#pragma unroll
            for (int j = 0; j < 8; ++j) {
                a[j]     = stage[wave][l16][cc0 + j];
                a[8 + j] = stage[wave][l16][cc1 + j];
            }
        }
        __syncthreads();
    }

    // Final layer 32 -> 3 (one WMMA; columns >= 3 use clamped index, their
    // results are simply never stored)
    {
        bool cv  = l16 < 3;
        int  ccl = cv ? l16 : 0;            // clamp: loads unconditional
        int  kb  = half_hi * 16;
        const half_t* w4 = &WT4[ccl][kb];   // contiguous 16 halves
        v16h b4;
#pragma unroll
        for (int j = 0; j < 16; ++j)
            b4[j] = w4[j];

        float bias = (float)Bias[3][ccl];
        v8f c4;
#pragma unroll
        for (int r = 0; r < 8; ++r) c4[r] = bias;

        c4 = __builtin_amdgcn_wmma_f32_16x16x32_f16(false, a, false, b4,
                                                    (short)0, c4, false, false);
        if (cv) {
#pragma unroll
            for (int r = 0; r < 8; ++r) {
                int row = half_hi * 8 + r;
                size_t p = (size_t)n * S_PTS + pt_base + row;
                out[p * 3 + l16] = c4[r];
            }
        }
    }
}

// ---------------------------------------------------------------------------
extern "C" void kernel_launch(void* const* d_in, const int* in_sizes, int n_in,
                              void* d_out, int out_size, void* d_ws, size_t ws_size,
                              hipStream_t stream)
{
    const float* x            = (const float*)d_in[0];
    const float* loc          = (const float*)d_in[1];
    const float* log_scale    = (const float*)d_in[2];
    const float* h_loc        = (const float*)d_in[3];
    const float* h_log_scale  = (const float*)d_in[4];
    const float* hh_loc       = (const float*)d_in[5];
    const float* hh_log_scale = (const float*)d_in[6];
    const float* eps          = (const float*)d_in[7];
    const float* h_eps        = (const float*)d_in[8];
    const float* hh_eps       = (const float*)d_in[9];
    const float* W_map        = (const float*)d_in[10];
    const float* b_map        = (const float*)d_in[11];
    const int*   h_gidx       = (const int*)d_in[12];
    const int*   hh_gidx      = (const int*)d_in[13];
    float* out = (float*)d_out;

    // workspace layout: z_h (96*4096 f16 = 768KB) | params (96*3267 f32 ~1.25MB)
    half_t* z_h   = (half_t*)d_ws;
    float*  pars  = (float*)((char*)d_ws + (1u << 20));

    z_kernel<<<(N_SAMP * P_DIM + 255) / 256, 256, 0, stream>>>(
        loc, log_scale, h_loc, h_log_scale, hh_loc, hh_log_scale,
        eps, h_eps, hh_eps, h_gidx, hh_gidx, z_h);

    int total_waves = 6 * NT_B;             // 1230 tile-waves
    params_gemm_kernel<<<(total_waves + 3) / 4, 128, 0, stream>>>(
        z_h, W_map, b_map, pars);

    mlp_kernel<<<dim3(32, N_SAMP), 256, 0, stream>>>(x, pars, out);
}
